// MultiHeadAttention_90271622627583
// MI455X (gfx1250) — compile-verified, matrix operand order checked by reading, not run on hardware
//
#include <hip/hip_runtime.h>

typedef __bf16 bf16_t;
typedef __attribute__((ext_vector_type(16))) __bf16 v16bf;
typedef __attribute__((ext_vector_type(8)))  float  v8f;

static constexpr int B = 2, S = 2048, D = 512, H = 8, E = 64;

union frag16 { v16bf v; uint4 u[2]; bf16_t h[16]; };
union pack8  { uint4 u; bf16_t h[8]; };

// Native hardware conversion (RNE), backend picks v_cvt for gfx1250.
__device__ __forceinline__ bf16_t f2bf(float x) { return (bf16_t)x; }

__device__ __forceinline__ v8f vzero8() {
    v8f v;
#pragma unroll
    for (int i = 0; i < 8; ++i) v[i] = 0.0f;
    return v;
}

// Load one 16x32 bf16 WMMA operand fragment (A or B style) for this lane.
// p points at (row-for-this-lane)*stride + k0; lanes 0-15 read halfs {0..7,16..23},
// lanes 16-31 read {8..15,24..31}  (hi8 = 8 for upper half-wave).
__device__ __forceinline__ v16bf load_frag(const bf16_t* p, int hi8) {
    frag16 f;
    f.u[0] = *reinterpret_cast<const uint4*>(p + hi8);
    f.u[1] = *reinterpret_cast<const uint4*>(p + 16 + hi8);
    return f.v;
}

__device__ __forceinline__ v8f wmma_bf16(v16bf a, v16bf b, v8f c) {
    return __builtin_amdgcn_wmma_f32_16x16x32_bf16(false, a, false, b, (short)0, c, false, false);
}

// ---------------- conversion / layout kernels ----------------

__global__ void cvt_f32_bf16(const float* __restrict__ in, bf16_t* __restrict__ out, int n) {
    int i = blockIdx.x * blockDim.x + threadIdx.x;
    if (i < n) out[i] = f2bf(in[i]);
}

// in[batch][rows][cols] -> out[batch][cols][rows] (bf16)
__global__ void cvt_transpose_w(const float* __restrict__ in, bf16_t* __restrict__ out,
                                int rows, int cols, int batch) {
    int i = blockIdx.x * blockDim.x + threadIdx.x;
    int tot = rows * cols * batch;
    if (i >= tot) return;
    int c = i % cols;
    int r = (i / cols) % rows;
    int bb = i / (rows * cols);
    out[((size_t)bb * cols + c) * rows + r] = f2bf(in[i]);
}

__global__ void cvt_mask_u8(const int* __restrict__ in, unsigned char* __restrict__ out, int n) {
    int i = blockIdx.x * blockDim.x + threadIdx.x;
    if (i < n) out[i] = (unsigned char)in[i];
}

// ---------------- QKV projection: q,k -> [BH,S,E], v -> [BH,E,S] (bf16) ----------------
// q is pre-scaled by 1/sqrt(E) = 0.125 so the attention inner loop needs no score scaling.

__global__ void __launch_bounds__(128) qkv_kernel(
    const bf16_t* __restrict__ xb,
    const bf16_t* __restrict__ wqt, const bf16_t* __restrict__ wkt, const bf16_t* __restrict__ wvt,
    const float* __restrict__ bq, const float* __restrict__ bk, const float* __restrict__ bv,
    bf16_t* __restrict__ qd, bf16_t* __restrict__ kd, bf16_t* __restrict__ vTd)
{
    const int lane = threadIdx.x & 31;
    const int wave = threadIdx.x >> 5;
    const int bh = blockIdx.y;
    const int b = bh >> 3, h = bh & 7;
    const int s0 = (blockIdx.x * 4 + wave) * 16;
    const int n = lane & 15;
    const int hi8 = (lane >> 4) << 3;

    v8f cq[4], ck[4], cv[4];
#pragma unroll
    for (int i = 0; i < 4; ++i) { cq[i] = vzero8(); ck[i] = vzero8(); cv[i] = vzero8(); }

    const bf16_t* xrow = xb + ((size_t)b * S + s0 + n) * D;   // A: rows of x
    const bf16_t* wqh = wqt + (size_t)h * E * D;              // [E,D] per head
    const bf16_t* wkh = wkt + (size_t)h * E * D;
    const bf16_t* wvh = wvt + (size_t)h * E * D;

    for (int d0 = 0; d0 < D; d0 += 32) {
        v16bf ax = load_frag(xrow + d0, hi8);
#pragma unroll
        for (int sub = 0; sub < 4; ++sub) {
            const size_t wr = (size_t)(sub * 16 + n) * D + d0; // B: col n = output e
            cq[sub] = wmma_bf16(ax, load_frag(wqh + wr, hi8), cq[sub]);
            ck[sub] = wmma_bf16(ax, load_frag(wkh + wr, hi8), ck[sub]);
            cv[sub] = wmma_bf16(ax, load_frag(wvh + wr, hi8), cv[sub]);
        }
    }

    bf16_t* qout = qd + (size_t)bh * S * E;
    bf16_t* kout = kd + (size_t)bh * S * E;
    bf16_t* vout = vTd + (size_t)bh * E * S;
#pragma unroll
    for (int sub = 0; sub < 4; ++sub) {
        const int e = sub * 16 + n;
        const float bqv = bq[h * E + e], bkv = bk[h * E + e], bvv = bv[h * E + e];
#pragma unroll
        for (int r = 0; r < 8; ++r) {
            const int s = s0 + hi8 + r;                 // C row = r + 8*(lane>=16)
            qout[(size_t)s * E + e] = f2bf((cq[sub][r] + bqv) * 0.125f);  // fold 1/sqrt(E)
            kout[(size_t)s * E + e] = f2bf(ck[sub][r] + bkv);
        }
        pack8 pv;
#pragma unroll
        for (int r = 0; r < 8; ++r) pv.h[r] = f2bf(cv[sub][r] + bvv);
        *reinterpret_cast<uint4*>(vout + (size_t)e * S + s0 + hi8) = pv.u; // vT[e, s0+hi8..+7]
    }
}

// ---------------- fused flash attention (transposed-score formulation) ----------------
// Per wave: 16 query columns; loop key blocks of 32. 8 waves/WG share the same (b,h)
// and key sequence so K/V stream is reused out of the WGP cache.
// S^T = K * q^T  (C cols = q -> lanes; softmax reductions are per-lane + one shfl_xor(16)).
// O^T = V^T * P^T (P^T B-fragment packs directly from S^T C registers, no LDS/transpose).

__global__ void __launch_bounds__(256) attn_kernel(
    const bf16_t* __restrict__ q, const bf16_t* __restrict__ k, const bf16_t* __restrict__ vT,
    const unsigned char* __restrict__ mask8, bf16_t* __restrict__ o_bf)
{
    const int lane = threadIdx.x & 31;
    const int wave = threadIdx.x >> 5;
    const int bh = blockIdx.y;
    const int b = bh >> 3, h = bh & 7;
    const int q0 = (blockIdx.x * 8 + wave) * 16;
    const int n = lane & 15;
    const int hi8 = (lane >> 4) << 3;
    const float L2E = 1.44269504088896f;

    const bf16_t* qrow = q + ((size_t)bh * S + q0 + n) * E;   // B operand: q^T col n
    const v16bf qb0 = load_frag(qrow, hi8);                   // e 0..31
    const v16bf qb1 = load_frag(qrow + 32, hi8);              // e 32..63

    const bf16_t* kbase = k + (size_t)bh * S * E;
    const bf16_t* vbase = vT + (size_t)bh * E * S;
    const unsigned char* mrow = mask8 + ((size_t)b * S + q0 + n) * S;

    v8f acc[4];
#pragma unroll
    for (int i = 0; i < 4; ++i) acc[i] = vzero8();
    float mrun = -3.0e38f, lrun = 0.0f;

    for (int t0 = 0; t0 < S; t0 += 32) {
        // ---- S^T tiles: rows t (2 tiles of 16), cols q ----
        const bf16_t* krA = kbase + (size_t)(t0 + n) * E;        // A rows t0..t0+15
        const bf16_t* krB = kbase + (size_t)(t0 + 16 + n) * E;   // A rows t0+16..t0+31
        v8f sta = vzero8(), stb = vzero8();
        sta = wmma_bf16(load_frag(krA, hi8), qb0, sta);
        sta = wmma_bf16(load_frag(krA + 32, hi8), qb1, sta);
        stb = wmma_bf16(load_frag(krB, hi8), qb0, stb);
        stb = wmma_bf16(load_frag(krB + 32, hi8), qb1, stb);

        if (t0 + 32 < S) {
            __builtin_prefetch(kbase + (size_t)(t0 + 32 + n) * E, 0, 3);
            __builtin_prefetch(mrow + t0 + 32, 0, 3);
        }

        // ---- mask (q already carries 1/sqrt(E)); block max per q column (== per lane) ----
        const unsigned long long mA = *reinterpret_cast<const unsigned long long*>(mrow + t0 + hi8);
        const unsigned long long mB = *reinterpret_cast<const unsigned long long*>(mrow + t0 + 16 + hi8);
        float sa[8], sb[8];
        float bmax = -3.0e38f;
#pragma unroll
        for (int r = 0; r < 8; ++r) {
            sa[r] = __builtin_fmaf(-1.0e9f, (float)((unsigned char)(mA >> (8 * r))), sta[r]);
            sb[r] = __builtin_fmaf(-1.0e9f, (float)((unsigned char)(mB >> (8 * r))), stb[r]);
            bmax = fmaxf(bmax, fmaxf(sa[r], sb[r]));
        }
        bmax = fmaxf(bmax, __shfl_xor(bmax, 16, 32));  // combine t-halves (same q column)
        const float mnew = fmaxf(mrun, bmax);
        const float rescale = exp2f((mrun - mnew) * L2E);
        lrun *= rescale;
#pragma unroll
        for (int i = 0; i < 4; ++i) acc[i] = acc[i] * rescale;

        // ---- P^T = exp(S^T - m); pack directly into B fragment ----
        frag16 pf;
        float bsum = 0.0f;
#pragma unroll
        for (int r = 0; r < 8; ++r) {
            const float pa = exp2f((sa[r] - mnew) * L2E);
            const float pb = exp2f((sb[r] - mnew) * L2E);
            bsum += pa + pb;
            pf.h[r] = f2bf(pa);       // K = t (lo run) from STa regs
            pf.h[8 + r] = f2bf(pb);   // K = t (hi run) from STb regs
        }
        bsum += __shfl_xor(bsum, 16, 32);
        lrun += bsum;
        mrun = mnew;

        // ---- O^T += V^T(e,t) * P^T(t,q) : 4 e-subtiles ----
#pragma unroll
        for (int sub = 0; sub < 4; ++sub) {
            const bf16_t* vrow = vbase + (size_t)(sub * 16 + n) * S + t0;
            acc[sub] = wmma_bf16(load_frag(vrow, hi8), pf.v, acc[sub]);
        }
    }

    // ---- normalize and store o (bf16, [B,S,D] with head offset) ----
    const float inv = 1.0f / lrun;
    bf16_t* orow = o_bf + ((size_t)b * S + q0 + n) * D + h * E;
#pragma unroll
    for (int sub = 0; sub < 4; ++sub) {
        pack8 po;
#pragma unroll
        for (int r = 0; r < 8; ++r) po.h[r] = f2bf(acc[sub][r] * inv);
        *reinterpret_cast<uint4*>(orow + sub * 16 + hi8) = po.u;
    }
}

// ---------------- output projection: out = o_bf @ Wo + bo (fp32 out) ----------------

__global__ void __launch_bounds__(128) oproj_kernel(
    const bf16_t* __restrict__ o_bf, const bf16_t* __restrict__ woT,
    const float* __restrict__ bo, float* __restrict__ out)
{
    const int lane = threadIdx.x & 31;
    const int wave = threadIdx.x >> 5;
    const int row0 = (blockIdx.x * 4 + wave) * 16;  // over B*S = 4096
    const int n0 = blockIdx.y * 64;                 // over D = 512
    const int n = lane & 15;
    const int hi8 = (lane >> 4) << 3;

    v8f c[4];
#pragma unroll
    for (int i = 0; i < 4; ++i) c[i] = vzero8();

    const bf16_t* arow = o_bf + (size_t)(row0 + n) * D;
    for (int d0 = 0; d0 < D; d0 += 32) {
        v16bf a = load_frag(arow + d0, hi8);
#pragma unroll
        for (int sub = 0; sub < 4; ++sub) {
            const bf16_t* wrow = woT + (size_t)(n0 + sub * 16 + n) * D + d0;
            c[sub] = wmma_bf16(a, load_frag(wrow, hi8), c[sub]);
        }
    }
#pragma unroll
    for (int sub = 0; sub < 4; ++sub) {
        const int col = n0 + sub * 16 + n;
        const float bias = bo[col];
#pragma unroll
        for (int r = 0; r < 8; ++r) {
            out[(size_t)(row0 + hi8 + r) * D + col] = c[sub][r] + bias;
        }
    }
}

// ---------------- host ----------------

extern "C" void kernel_launch(void* const* d_in, const int* in_sizes, int n_in,
                              void* d_out, int out_size, void* d_ws, size_t ws_size,
                              hipStream_t stream) {
    const float* x  = (const float*)d_in[0];
    const int* am   = (const int*)d_in[1];
    const float* Wq = (const float*)d_in[2];
    const float* bq = (const float*)d_in[3];
    const float* Wk = (const float*)d_in[4];
    const float* bk = (const float*)d_in[5];
    const float* Wv = (const float*)d_in[6];
    const float* bv = (const float*)d_in[7];
    const float* Wo = (const float*)d_in[8];
    const float* bo = (const float*)d_in[9];
    float* out = (float*)d_out;

    char* ws = (char*)d_ws;
    size_t off = 0;
    auto take = [&](size_t bytes) -> char* {
        char* p = ws + off;
        off = (off + bytes + 255) & ~(size_t)255;
        return p;
    };
    bf16_t* xb  = (bf16_t*)take((size_t)B * S * D * 2);
    bf16_t* wqt = (bf16_t*)take((size_t)H * E * D * 2);
    bf16_t* wkt = (bf16_t*)take((size_t)H * E * D * 2);
    bf16_t* wvt = (bf16_t*)take((size_t)H * E * D * 2);
    bf16_t* wot = (bf16_t*)take((size_t)D * D * 2);
    bf16_t* qb  = (bf16_t*)take((size_t)B * H * S * E * 2);
    bf16_t* kb  = (bf16_t*)take((size_t)B * H * S * E * 2);
    bf16_t* vtb = (bf16_t*)take((size_t)B * H * S * E * 2);
    unsigned char* m8 = (unsigned char*)take((size_t)B * S * S);
    bf16_t* ob  = (bf16_t*)take((size_t)B * S * D * 2);

    const int nx = B * S * D;
    cvt_f32_bf16<<<(nx + 255) / 256, 256, 0, stream>>>(x, xb, nx);
    const int nw = H * D * E;
    cvt_transpose_w<<<(nw + 255) / 256, 256, 0, stream>>>(Wq, wqt, D, E, H);
    cvt_transpose_w<<<(nw + 255) / 256, 256, 0, stream>>>(Wk, wkt, D, E, H);
    cvt_transpose_w<<<(nw + 255) / 256, 256, 0, stream>>>(Wv, wvt, D, E, H);
    const int nwo = D * D;
    cvt_transpose_w<<<(nwo + 255) / 256, 256, 0, stream>>>(Wo, wot, D, D, 1);
    const int nm = B * S * S;
    cvt_mask_u8<<<(nm + 255) / 256, 256, 0, stream>>>(am, m8, nm);

    qkv_kernel<<<dim3(S / 64, B * H), 128, 0, stream>>>(xb, wqt, wkt, wvt, bq, bk, bv, qb, kb, vtb);
    attn_kernel<<<dim3(S / 128, B * H), 256, 0, stream>>>(qb, kb, vtb, m8, ob);
    oproj_kernel<<<dim3((B * S) / 64, D / 64), 128, 0, stream>>>(ob, wot, bo, out);
}
